// SConv_2d_33706903339140
// MI455X (gfx1250) — matrix operand order repaired and checked by reading, not compiled, not run on hardware
//
#include <hip/hip_runtime.h>
#include <hip/hip_bf16.h>
#include <math.h>

typedef __attribute__((ext_vector_type(16))) __bf16 v16bf;
typedef __attribute__((ext_vector_type(8)))  float  v8f;

#define CIN   64
#define COUT  128
#define HW    128
#define KTOT  576              // 64 channels * 9 taps
#define ZSTR  580              // ushort stride per pixel row (576 + pad for banks)
#define NPIX  64               // pixels per workgroup
#define NTHR  128              // 4 waves
#define NKC   18               // K chunks: 576 / 32
#define NM    8                // M tiles: 128 / 16
#define WFRAG_ELEMS (NM * NKC * 32 * 16)          // ushort count
#define WFRAG_BYTES (WFRAG_ELEMS * 2)             // 147456 B

// f32 -> bf16, round-to-nearest-even (bit-level)
__device__ __forceinline__ unsigned short f2bf(float f) {
    unsigned u = __builtin_bit_cast(unsigned, f);
    u += 0x7fffu + ((u >> 16) & 1u);
    return (unsigned short)(u >> 16);
}

#define CSWAP(a, b) { float _lo = fminf(a, b); float _hi = fmaxf(a, b); a = _lo; b = _hi; }

// Load one A fragment (row M = 16m + lane%16) from f32 W with on-the-fly cvt.
// ISA 16-bit A layout: elements 0..7 = K k0+8*khalf..+7, 8..15 = K k0+16+8*khalf..+23.
__device__ __forceinline__ v16bf loadA_cvt(const float* __restrict__ wr, int k0, int khalf) {
    const float* wp = wr + k0 + 8 * khalf;
    float4 a0 = *(const float4*)(wp);
    float4 a1 = *(const float4*)(wp + 4);
    float4 a2 = *(const float4*)(wp + 16);
    float4 a3 = *(const float4*)(wp + 20);
    union { v16bf v; unsigned short e[16]; } A;
    A.e[0]  = f2bf(a0.x); A.e[1]  = f2bf(a0.y); A.e[2]  = f2bf(a0.z); A.e[3]  = f2bf(a0.w);
    A.e[4]  = f2bf(a1.x); A.e[5]  = f2bf(a1.y); A.e[6]  = f2bf(a1.z); A.e[7]  = f2bf(a1.w);
    A.e[8]  = f2bf(a2.x); A.e[9]  = f2bf(a2.y); A.e[10] = f2bf(a2.z); A.e[11] = f2bf(a2.w);
    A.e[12] = f2bf(a3.x); A.e[13] = f2bf(a3.y); A.e[14] = f2bf(a3.z); A.e[15] = f2bf(a3.w);
    return A.v;
}

// Prologue: pack W (f32, [o][k] with k = c*9+t) into per-lane WMMA A fragments
// (bf16) at ws[((m*18 + kc)*32 + lane)*16 ..]. One thread per fragment-lane.
__global__ __launch_bounds__(256) void pack_w(const float* __restrict__ wgt,
                                              unsigned short* __restrict__ wfrag) {
    int t = blockIdx.x * 256 + threadIdx.x;       // 0 .. 4607
    if (t >= NM * NKC * 32) return;
    int lane  = t & 31;
    int kc    = (t >> 5) % NKC;
    int m     = (t >> 5) / NKC;
    int nloc  = lane & 15;
    int khalf = lane >> 4;
    const float* wr = wgt + (m * 16 + nloc) * KTOT;
    union { v16bf v; uint4 q[2]; } A;
    A.v = loadA_cvt(wr, kc * 32, khalf);
    uint4* dst = (uint4*)(wfrag + t * 16);        // 32B aligned
    dst[0] = A.q[0];
    dst[1] = A.q[1];
}

template <bool PACKED>
__global__ __launch_bounds__(NTHR) void sconv_fused(
        const float* __restrict__ x,
        const float* __restrict__ coef,
        const float* __restrict__ wgt,
        const float* __restrict__ bias,
        const unsigned short* __restrict__ wfrag,
        float* __restrict__ out)
{
    __shared__ float          xt[CIN * 3 * 66];   // 50688 B: 3 halo rows x 66 cols x 64 ch
    __shared__ unsigned short zl[NPIX * ZSTR];    // 74240 B: z in bf16, [pixel][k]

    const int tid  = threadIdx.x;
    const int blk  = blockIdx.x;
    const int wseg = blk & 1;                 // which 64-pixel half of the row
    const int h    = (blk >> 1) & (HW - 1);   // output row
    const int bb   = blk >> 8;                // batch
    const int w0   = wseg * NPIX;

    // ---- stage x tile (with halo, zero padded) into LDS ----
    for (int f = tid; f < CIN * 3 * 66; f += NTHR) {
        int c   = f / 198;
        int rem = f - c * 198;
        int di  = rem / 66;
        int ww  = rem - di * 66;
        int row = h - 1 + di;
        int col = w0 - 1 + ww;
        float v = 0.f;
        if (row >= 0 && row < HW && col >= 0 && col < HW)
            v = x[(((bb * CIN + c) * HW) + row) * HW + col];
        xt[f] = v;
    }

    // coefficient rows for the 8 non-center outputs, kept in VGPRs (L2-hot loads)
    float cr[8][9];
    #pragma unroll
    for (int ii = 0; ii < 8; ++ii) {
        int i = (ii < 4) ? ii : ii + 1;       // skip center row 4
        #pragma unroll
        for (int j = 0; j < 9; ++j) cr[ii][j] = coef[i * 9 + j];
    }

    __syncthreads();

    // ---- Phase A: smooth (8x9 FMA) + 8-way Batcher sort -> z (bf16) in LDS ----
    for (int it = 0; it < (CIN * NPIX) / NTHR; ++it) {
        int idx = it * NTHR + tid;
        int c = idx >> 6;
        int n = idx & 63;
        const float* xr = &xt[c * 198 + n];
        float p[9];
        #pragma unroll
        for (int di = 0; di < 3; ++di)
            #pragma unroll
            for (int dj = 0; dj < 3; ++dj)
                p[di * 3 + dj] = xr[di * 66 + dj];

        float y[8];
        #pragma unroll
        for (int ii = 0; ii < 8; ++ii) {
            float a = cr[ii][0] * p[0];
            #pragma unroll
            for (int j = 1; j < 9; ++j) a = fmaf(cr[ii][j], p[j], a);
            y[ii] = a;
        }

        // Batcher odd-even mergesort, 19 compare-exchanges, ascending
        CSWAP(y[0], y[1]) CSWAP(y[2], y[3]) CSWAP(y[4], y[5]) CSWAP(y[6], y[7])
        CSWAP(y[0], y[2]) CSWAP(y[1], y[3]) CSWAP(y[4], y[6]) CSWAP(y[5], y[7])
        CSWAP(y[1], y[2]) CSWAP(y[5], y[6])
        CSWAP(y[0], y[4]) CSWAP(y[1], y[5]) CSWAP(y[2], y[6]) CSWAP(y[3], y[7])
        CSWAP(y[2], y[4]) CSWAP(y[3], y[5])
        CSWAP(y[1], y[2]) CSWAP(y[3], y[4]) CSWAP(y[5], y[6])

        // z = [s0..s3, raw center, s4..s7]  at k = c*9 + t
        unsigned short* zr = &zl[n * ZSTR + c * 9];
        zr[0] = f2bf(y[0]); zr[1] = f2bf(y[1]); zr[2] = f2bf(y[2]); zr[3] = f2bf(y[3]);
        zr[4] = f2bf(p[4]);
        zr[5] = f2bf(y[4]); zr[6] = f2bf(y[5]); zr[7] = f2bf(y[6]); zr[8] = f2bf(y[7]);
    }

    __syncthreads();

    // ---- Phase B: out[o,pix] = b[o] + sum_k Wbf[o,k] * z[k,pix] via WMMA bf16 ----
    const int lane  = tid & 31;
    const int wv    = tid >> 5;       // wave id: 16-pixel N tile
    const int nloc  = lane & 15;
    const int khalf = lane >> 4;
    const int pix   = wv * 16 + nloc;
    const unsigned short* zr = &zl[pix * ZSTR];
    const int wglob = w0 + pix;

    for (int mp = 0; mp < 4; ++mp) {          // two 16-row M tiles per pass
        const int m0 = mp * 2, m1 = mp * 2 + 1;
        v8f acc0 = {0.f, 0.f, 0.f, 0.f, 0.f, 0.f, 0.f, 0.f};
        v8f acc1 = {0.f, 0.f, 0.f, 0.f, 0.f, 0.f, 0.f, 0.f};

        // packed-fragment base pointers (uint4 units): ((m*18+kc)*32+lane)*2
        const uint4* fr0 = (const uint4*)wfrag + (m0 * NKC * 32 + lane) * 2;
        const uint4* fr1 = (const uint4*)wfrag + (m1 * NKC * 32 + lane) * 2;
        // fallback f32 row pointers
        const float* wr0 = wgt + (m0 * 16 + nloc) * KTOT;
        const float* wr1 = wgt + (m1 * 16 + nloc) * KTOT;

        for (int kc = 0; kc < NKC; ++kc) {    // K = 576 = 18 * 32
            const int k0 = kc * 32;
            // B fragment from LDS z (col N=lane%16): elements 0..7 = K k0+8*khalf..+7,
            // 8..15 = K k0+16+8*khalf..+23 (same order as A).
            union { v16bf v; uint2 q[4]; } B;
            const uint2* bp = (const uint2*)(zr + k0 + 8 * khalf);
            B.q[0] = bp[0]; B.q[1] = bp[1]; B.q[2] = bp[4]; B.q[3] = bp[5];

            v16bf A0, A1;
            if (PACKED) {
                union { v16bf v; uint4 q[2]; } U0, U1;
                U0.q[0] = fr0[kc * 64];  U0.q[1] = fr0[kc * 64 + 1];
                U1.q[0] = fr1[kc * 64];  U1.q[1] = fr1[kc * 64 + 1];
                A0 = U0.v; A1 = U1.v;
            } else {
                A0 = loadA_cvt(wr0, k0, khalf);
                A1 = loadA_cvt(wr1, k0, khalf);
            }

            // two independent accumulation chains sharing one B fragment
            acc0 = __builtin_amdgcn_wmma_f32_16x16x32_bf16(
                       false, A0, false, B.v, (short)0, acc0, false, false);
            acc1 = __builtin_amdgcn_wmma_f32_16x16x32_bf16(
                       false, A1, false, B.v, (short)0, acc1, false, false);
        }

        // C/D layout: VGPR r -> M = r + 8*khalf, N = lane%16
        #pragma unroll
        for (int r = 0; r < 8; ++r) {
            int o0 = m0 * 16 + r + 8 * khalf;
            int o1 = m1 * 16 + r + 8 * khalf;
            out[(((bb * COUT + o0) * HW) + h) * HW + wglob] = acc0[r] + bias[o0];
            out[(((bb * COUT + o1) * HW) + h) * HW + wglob] = acc1[r] + bias[o1];
        }
    }
}

extern "C" void kernel_launch(void* const* d_in, const int* in_sizes, int n_in,
                              void* d_out, int out_size, void* d_ws, size_t ws_size,
                              hipStream_t stream) {
    (void)in_sizes; (void)n_in; (void)out_size;
    const float* x    = (const float*)d_in[0];   // (8,64,128,128)
    const float* coef = (const float*)d_in[1];   // (9,9)
    const float* wgt  = (const float*)d_in[2];   // (128,64,3,3)
    const float* bias = (const float*)d_in[3];   // (128,)
    float* out        = (float*)d_out;           // (8,128,128,128)

    dim3 grid(8 * 128 * 2);   // batch * rows * 2 half-rows
    dim3 block(NTHR);

    if (ws_size >= (size_t)WFRAG_BYTES && d_ws != nullptr) {
        unsigned short* wfrag = (unsigned short*)d_ws;
        pack_w<<<dim3((NM * NKC * 32 + 255) / 256), dim3(256), 0, stream>>>(wgt, wfrag);
        sconv_fused<true><<<grid, block, 0, stream>>>(x, coef, wgt, bias, wfrag, out);
    } else {
        sconv_fused<false><<<grid, block, 0, stream>>>(x, coef, wgt, bias, nullptr, out);
    }
}
